// AttentionFlow_60275571032360
// MI455X (gfx1250) — matrix operand order repaired and checked by reading, not compiled, not run on hardware
//
#include <hip/hip_runtime.h>

typedef __attribute__((ext_vector_type(16))) __bf16 bf16x16;
typedef __attribute__((ext_vector_type(8)))  __bf16 bf16x8;
typedef __attribute__((ext_vector_type(8)))  float  floatx8;
typedef __attribute__((ext_vector_type(4)))  int    v4i;

constexpr int kB = 16, kN = 1024, kM = 128, kD = 768;
constexpr float kNeg = -1e9f;

#if defined(__has_builtin)
#if __has_builtin(__builtin_amdgcn_global_load_async_to_lds_b128) && \
    __has_builtin(__builtin_amdgcn_s_wait_asynccnt)
#define USE_ASYNC_LDS 1
#endif
#endif

// 16-byte global -> LDS copy (async on gfx1250 toolchains that expose it)
__device__ __forceinline__ void cp16_g2l(const __bf16* g, __bf16* l) {
#ifdef USE_ASYNC_LDS
  __builtin_amdgcn_global_load_async_to_lds_b128(
      (__attribute__((address_space(1))) v4i*)(g),
      (__attribute__((address_space(3))) v4i*)(l), 0, 0);
#else
  *(float4*)l = *(const float4*)g;
#endif
}
__device__ __forceinline__ void wait_async_lds() {
#ifdef USE_ASYNC_LDS
  __builtin_amdgcn_s_wait_asynccnt(0);
#endif
}

// lane exchange via ds_swizzle (xor pattern), avoids bpermute address setup
template <int XM>
__device__ __forceinline__ float sw_xor(float v) {
  int r = __builtin_amdgcn_ds_swizzle(__float_as_int(v), (XM << 5) | 0x1f);
  return __int_as_float(r);
}
// reductions within each 16-lane half of a wave32
__device__ __forceinline__ float half_max(float v) {
  v = fmaxf(v, sw_xor<1>(v)); v = fmaxf(v, sw_xor<2>(v));
  v = fmaxf(v, sw_xor<4>(v)); v = fmaxf(v, sw_xor<8>(v));
  return v;
}
__device__ __forceinline__ float half_sum(float v) {
  v += sw_xor<1>(v); v += sw_xor<2>(v); v += sw_xor<4>(v); v += sw_xor<8>(v);
  return v;
}
__device__ __forceinline__ float wave_sum(float v) {
  v = half_sum(v); v += sw_xor<16>(v);
  return v;
}

__device__ __forceinline__ bf16x16 join8(bf16x8 lo, bf16x8 hi) {
  bf16x16 r;
#pragma unroll
  for (int e = 0; e < 8; ++e) { r[e] = lo[e]; r[8 + e] = hi[e]; }
  return r;
}

// ---------------- prep: c_logit = C.w1 ; Cw3b = bf16(C*w3) ----------------
__global__ __launch_bounds__(256) void prep_c_kernel(
    const float* __restrict__ C, const float* __restrict__ w1,
    const float* __restrict__ w3, __bf16* __restrict__ Cw3b,
    float* __restrict__ c_logit) {
  const int wave = threadIdx.x >> 5, lane = threadIdx.x & 31;
  const int row = blockIdx.x * 8 + wave;               // row in [0, B*N)
  const float* src = C + (size_t)row * kD;
  __bf16* dst = Cw3b + (size_t)row * kD;
  float s = 0.f;
  for (int k = lane; k < kD; k += 32) {
    float x = src[k];
    s += x * w1[k];
    dst[k] = (__bf16)(x * w3[k]);
  }
  s = wave_sum(s);
  if (lane == 0) c_logit[row] = s;
}

// -------- prep: q_logit = Q.w2 ; Qb row-major bf16 ; QbT = [d x m] bf16 ----
__global__ __launch_bounds__(256) void prep_q_kernel(
    const float* __restrict__ Q, const float* __restrict__ w2,
    __bf16* __restrict__ Qb, __bf16* __restrict__ QbT,
    float* __restrict__ q_logit) {
  const int wave = threadIdx.x >> 5, lane = threadIdx.x & 31;
  const int row = blockIdx.x * 8 + wave;               // row = b*M + j
  const int b = row / kM, j = row % kM;
  const float* src = Q + (size_t)row * kD;
  __bf16* dst = Qb + (size_t)row * kD;
  __bf16* dstT = QbT + (size_t)b * kD * kM + j;
  float s = 0.f;
  for (int k = lane; k < kD; k += 32) {
    float x = src[k];
    s += x * w2[k];
    __bf16 xb = (__bf16)x;
    dst[k] = xb;
    dstT[(size_t)k * kM] = xb;
  }
  s = wave_sum(s);
  if (lane == 0) q_logit[row] = s;
}

// ---- fused: S tile (WMMA) -> masked softmax -> A tile (WMMA) -> out[0:3d) ----
constexpr int kAP = kD + 8;   // padded LDS row stride: 388 dwords -> 4-bank lane stride
__global__ __launch_bounds__(256) void attn_tile_kernel(
    const __bf16* __restrict__ Cw3b, const __bf16* __restrict__ Qb,
    const __bf16* __restrict__ QbT, const float* __restrict__ C,
    const float* __restrict__ c_logit, const float* __restrict__ q_logit,
    const unsigned char* __restrict__ c_mask,
    const unsigned char* __restrict__ q_mask,
    float* __restrict__ out, float* __restrict__ q2c) {
  const int b = blockIdx.y;
  const int i0 = blockIdx.x * 16;
  const int wave = threadIdx.x >> 5;
  const int lane = threadIdx.x & 31;
  const int sub = lane >> 4;       // which 16-lane half
  const int lh = lane & 15;

  __shared__ __bf16 sA[16][kAP];      // (C*w3) strip, bank-conflict-free pad
  __shared__ float redA[8][16];
  __shared__ float redB[8][16];
  __shared__ float rowMax[16];
  __shared__ float rowSum[16];
  __shared__ __bf16 sP[16][kM + 8];   // C2Q weights (bf16)

  // ---- stage A strip (16 x D bf16) into LDS with async global->LDS copies
  {
    const __bf16* Abase = Cw3b + ((size_t)b * kN + i0) * kD;
#pragma unroll
    for (int it = 0; it < 6; ++it) {               // 1536 x 16B chunks / 256 thr
      const int id = threadIdx.x + 256 * it;
      const int row = id / 96, kc = (id % 96) * 8; // 96 chunks per 768-elem row
      cp16_g2l(Abase + (size_t)row * kD + kc, &sA[row][kc]);
    }
  }
  // prefetch GEMM2 B rows (first tile of this wave) and C rows for the epilogue
  {
    const __bf16* pf = QbT + ((size_t)b * kD + wave * 16 + lh) * kM;
    __builtin_prefetch(pf, 0, 1);
    __builtin_prefetch(pf + 64, 0, 1);
    const float* pc = C + ((size_t)b * kN + i0 + lh) * kD + wave * 96;
    __builtin_prefetch(pc, 0, 1);
  }
  wait_async_lds();
  __syncthreads();

  // ---- GEMM1: S[16x128] = (C*w3)[16 x D] x Q^T[D x 128]; wave owns cols 16w..
  const __bf16* Brow = Qb + ((size_t)b * kM + wave * 16 + lh) * kD;   // N = j
  floatx8 acc = {};
#pragma unroll 4
  for (int k0 = 0; k0 < kD; k0 += 32) {
    bf16x8 alo = *(const bf16x8*)&sA[lh][k0 + 8 * sub];
    bf16x8 ahi = *(const bf16x8*)&sA[lh][k0 + 16 + 8 * sub];
    bf16x16 a = join8(alo, ahi);
    bf16x16 bb = join8(*(const bf16x8*)(Brow + k0 + 8 * sub),
                       *(const bf16x8*)(Brow + k0 + 16 + 8 * sub));
    acc = __builtin_amdgcn_wmma_f32_16x16x32_bf16(false, a, false, bb,
                                                  (short)0, acc, false, false);
  }

  const int j = wave * 16 + lh;
  const float ql = q_logit[b * kM + j];
  const bool qm = q_mask[b * kM + j] != 0;

  float S[8], Sm[8];
#pragma unroll
  for (int r = 0; r < 8; ++r) {
    const int i = i0 + r + 8 * sub;                 // D-matrix row M
    float s = acc[r] + c_logit[b * kN + i] + ql;
    const bool cm = c_mask[b * kN + i] != 0;
    S[r] = s;
    Sm[r] = (qm || cm) ? kNeg : s;
  }
  // per-(wave,row) partial max (masked + raw) via half-wave swizzles
#pragma unroll
  for (int r = 0; r < 8; ++r) {
    float mM = half_max(Sm[r]);
    float mR = half_max(S[r]);
    if (lh == 0) { redA[wave][r + 8 * sub] = mM; redB[wave][r + 8 * sub] = mR; }
  }
  __syncthreads();
  if (threadIdx.x < 16) {
    const int t = threadIdx.x;
    float mM = redA[0][t], mR = redB[0][t];
#pragma unroll
    for (int w = 1; w < 8; ++w) { mM = fmaxf(mM, redA[w][t]); mR = fmaxf(mR, redB[w][t]); }
    rowMax[t] = mM;
    const int i = i0 + t;
    q2c[b * kN + i] = (c_mask[b * kN + i] != 0) ? kNeg : mR;  // Q2C logits
  }
  __syncthreads();
  float p[8];
#pragma unroll
  for (int r = 0; r < 8; ++r) {
    p[r] = __expf(Sm[r] - rowMax[r + 8 * sub]);
    float s = half_sum(p[r]);
    if (lh == 0) redA[wave][r + 8 * sub] = s;
  }
  __syncthreads();
  if (threadIdx.x < 16) {
    const int t = threadIdx.x;
    float s = 0.f;
#pragma unroll
    for (int w = 0; w < 8; ++w) s += redA[w][t];
    rowSum[t] = s;
  }
  __syncthreads();
#pragma unroll
  for (int r = 0; r < 8; ++r) {
    const float P = p[r] / rowSum[r + 8 * sub];
    sP[r + 8 * sub][j] = (__bf16)P;                 // C2Q weights as bf16
  }
  __syncthreads();

  // ---- GEMM2: A[16 x D] = P[16 x 128] x Q[128 x D]; wave owns d-tiles w,w+8,...
  for (int t = wave; t < kD / 16; t += 8) {
    const int n0 = t * 16;
    const __bf16* BT = QbT + ((size_t)b * kD + n0 + lh) * kM;  // N = dcol
    floatx8 acc2 = {};
#pragma unroll
    for (int kk = 0; kk < kM; kk += 32) {
      bf16x8 alo = *(const bf16x8*)&sP[lh][kk + 8 * sub];
      bf16x8 ahi = *(const bf16x8*)&sP[lh][kk + 16 + 8 * sub];
      bf16x16 a = join8(alo, ahi);
      bf16x16 bb = join8(*(const bf16x8*)(BT + kk + 8 * sub),
                         *(const bf16x8*)(BT + kk + 16 + 8 * sub));
      acc2 = __builtin_amdgcn_wmma_f32_16x16x32_bf16(false, a, false, bb,
                                                     (short)0, acc2, false, false);
    }
#pragma unroll
    for (int r = 0; r < 8; ++r) {
      const int i = i0 + r + 8 * sub;
      const int col = n0 + lh;
      const float cv = C[((size_t)b * kN + i) * kD + col];
      const float av = acc2[r];
      float* o = out + ((size_t)b * kN + i) * (4 * kD);
      o[col] = cv;                 // out[:, 0:d)   = C
      o[kD + col] = av;            // out[:, d:2d)  = A
      o[2 * kD + col] = cv * av;   // out[:, 2d:3d) = C*A
    }
  }
}

// ---- per-batch: Q2C softmax over n, then Bctx = Q2C . C  (1xN x NxD) ----
__global__ __launch_bounds__(256) void q2c_bctx_kernel(
    const float* __restrict__ q2c, const float* __restrict__ C,
    float* __restrict__ Bctx) {
  const int b = blockIdx.x;
  const int tid = threadIdx.x;
  __shared__ float wq[kN];
  __shared__ float red[256];

  float lm = -3.4e38f;
  for (int i = tid; i < kN; i += 256) lm = fmaxf(lm, q2c[b * kN + i]);
  red[tid] = lm; __syncthreads();
  for (int s = 128; s > 0; s >>= 1) {
    if (tid < s) red[tid] = fmaxf(red[tid], red[tid + s]);
    __syncthreads();
  }
  const float gmax = red[0];
  __syncthreads();
  float ls = 0.f;
  for (int i = tid; i < kN; i += 256) {
    float e = __expf(q2c[b * kN + i] - gmax);
    wq[i] = e; ls += e;
  }
  red[tid] = ls; __syncthreads();
  for (int s = 128; s > 0; s >>= 1) {
    if (tid < s) red[tid] += red[tid + s];
    __syncthreads();
  }
  const float inv = 1.0f / red[0];
  __syncthreads();

  for (int col = tid; col < kD; col += 256) {
    float acc = 0.f;
    const float* cp = C + (size_t)b * kN * kD + col;
#pragma unroll 4
    for (int i = 0; i < kN; ++i) acc += wq[i] * cp[(size_t)i * kD];
    Bctx[b * kD + col] = acc * inv;
  }
}

// ---- elementwise: out[:, 3d:4d) = C * Bctx (broadcast over rows) ----
__global__ __launch_bounds__(256) void cxb_kernel(
    const float* __restrict__ C, const float* __restrict__ Bctx,
    float* __restrict__ out) {
  const size_t idx = (size_t)blockIdx.x * 256 + threadIdx.x;  // over B*N*(D/4)
  const size_t row = idx / (kD / 4);
  const int c4 = (int)(idx % (kD / 4));
  const int b = (int)(row / kN);
  const float4 c = ((const float4*)C)[idx];
  const float4 g = ((const float4*)(Bctx + (size_t)b * kD))[c4];
  float4 r;
  r.x = c.x * g.x; r.y = c.y * g.y; r.z = c.z * g.z; r.w = c.w * g.w;
  ((float4*)out)[row * kD + (3 * kD) / 4 + c4] = r;
}

extern "C" void kernel_launch(void* const* d_in, const int* in_sizes, int n_in,
                              void* d_out, int out_size, void* d_ws, size_t ws_size,
                              hipStream_t stream) {
  (void)in_sizes; (void)n_in; (void)out_size; (void)ws_size;
  const float* C = (const float*)d_in[0];
  const float* Q = (const float*)d_in[1];
  const unsigned char* c_mask = (const unsigned char*)d_in[2];
  const unsigned char* q_mask = (const unsigned char*)d_in[3];
  const float* w1 = (const float*)d_in[4];
  const float* w2 = (const float*)d_in[5];
  const float* w3 = (const float*)d_in[6];
  float* out = (float*)d_out;

  char* p = (char*)d_ws;
  auto alloc = [&](size_t bytes) -> char* {
    char* r = p; p += (bytes + 255) & ~(size_t)255; return r;
  };
  __bf16* Cw3b = (__bf16*)alloc((size_t)kB * kN * kD * 2);
  __bf16* Qb   = (__bf16*)alloc((size_t)kB * kM * kD * 2);
  __bf16* QbT  = (__bf16*)alloc((size_t)kB * kD * kM * 2);
  float* c_logit = (float*)alloc((size_t)kB * kN * 4);
  float* q_logit = (float*)alloc((size_t)kB * kM * 4);
  float* q2c     = (float*)alloc((size_t)kB * kN * 4);
  float* Bctx    = (float*)alloc((size_t)kB * kD * 4);

  prep_c_kernel<<<(kB * kN) / 8, 256, 0, stream>>>(C, w1, w3, Cw3b, c_logit);
  prep_q_kernel<<<(kB * kM) / 8, 256, 0, stream>>>(Q, w2, Qb, QbT, q_logit);
  attn_tile_kernel<<<dim3(kN / 16, kB), 256, 0, stream>>>(
      Cw3b, Qb, QbT, C, c_logit, q_logit, c_mask, q_mask, out, q2c);
  q2c_bctx_kernel<<<kB, 256, 0, stream>>>(q2c, C, Bctx);
  cxb_kernel<<<(kB * kN * (kD / 4)) / 256, 256, 0, stream>>>(C, Bctx, out);
}
